// GNN_18605798326743
// MI455X (gfx1250) — matrix operand order repaired and checked by reading, not compiled, not run on hardware
//
#include <hip/hip_runtime.h>

// ---------------------------------------------------------------------------
// 3-layer GCN for gfx1250 (MI455X).
// GEMMs via v_wmma_f32_16x16x32_bf16 (bf16 inputs, fp32 accumulate).
// Edge aggregation via per-launch CSR build (counting sort by dst) + pure
// gather-reduce: all feature matrices are L2-resident (102MB < 192MB L2),
// so random row gathers stay on-die and the hot phase needs no atomics.
// ---------------------------------------------------------------------------

typedef __bf16 bf16_t;
typedef bf16_t v16bf __attribute__((ext_vector_type(16)));
typedef float  v8f   __attribute__((ext_vector_type(8)));

__device__ __forceinline__ unsigned short f2bf(float f) {
  unsigned int u = __float_as_uint(f);
  u += 0x7FFFu + ((u >> 16) & 1u);     // round-to-nearest-even
  return (unsigned short)(u >> 16);
}

union Frag16 { v16bf v; uint4 q[2]; };

// --------------------------- WMMA GEMM -------------------------------------
// C[M,N] = A[M,K] @ B[K,N]; A,B,C fp32 in memory; bf16 in the MACs.
// Block tile 64x64, 128 threads (4 waves), each wave computes 32x32 (2x2 WMMA).
__global__ __launch_bounds__(128) void gcn_gemm_wmma(
    const float* __restrict__ A, const float* __restrict__ B,
    float* __restrict__ C, int M, int K, int N)
{
  __shared__ unsigned short ldsA[64 * 32];   // [row][k]
  __shared__ unsigned short ldsB[64 * 32];   // transposed: [n][k]

  const int tid   = threadIdx.x;
  const int lane  = tid & 31;
  const int wave  = tid >> 5;
  const int waveM = wave >> 1;   // 0..1
  const int waveN = wave & 1;    // 0..1
  const int half  = lane >> 4;   // 0/1
  const int ln    = lane & 15;

  const int rowBase = blockIdx.x * 64;
  const int colBase = blockIdx.y * 64;

  v8f zero = {0.f, 0.f, 0.f, 0.f, 0.f, 0.f, 0.f, 0.f};
  v8f acc[2][2];
  acc[0][0] = zero; acc[0][1] = zero; acc[1][0] = zero; acc[1][1] = zero;

  const int kTiles = K >> 5;
  for (int kt = 0; kt < kTiles; ++kt) {
    const int kBase = kt << 5;

    // ---- load A tile 64x32 fp32 -> bf16 LDS (row-major) ----
    {
      const int r  = tid >> 1;        // 0..63
      const int sg = tid & 1;         // half-row of 16 floats
      const int gr = rowBase + r;
      unsigned short tmp[16];
      if (gr < M) {
        const float4* p = (const float4*)&A[(size_t)gr * K + kBase + sg * 16];
        #pragma unroll
        for (int q = 0; q < 4; ++q) {
          float4 f = p[q];
          tmp[q * 4 + 0] = f2bf(f.x); tmp[q * 4 + 1] = f2bf(f.y);
          tmp[q * 4 + 2] = f2bf(f.z); tmp[q * 4 + 3] = f2bf(f.w);
        }
      } else {
        #pragma unroll
        for (int q = 0; q < 16; ++q) tmp[q] = 0;
      }
      uint4* dst = (uint4*)&ldsA[r * 32 + sg * 16];
      dst[0] = ((const uint4*)tmp)[0];
      dst[1] = ((const uint4*)tmp)[1];
    }

    // ---- load B tile 32x64 fp32 -> bf16 LDS transposed [n][k] ----
    {
      const int k  = tid >> 2;        // 0..31
      const int ns = tid & 3;         // 16 columns each
      const float4* p = (const float4*)&B[(size_t)(kBase + k) * N + colBase + ns * 16];
      #pragma unroll
      for (int q = 0; q < 4; ++q) {
        float4 f = p[q];
        int n0 = ns * 16 + q * 4;
        ldsB[(n0 + 0) * 32 + k] = f2bf(f.x);
        ldsB[(n0 + 1) * 32 + k] = f2bf(f.y);
        ldsB[(n0 + 2) * 32 + k] = f2bf(f.z);
        ldsB[(n0 + 3) * 32 + k] = f2bf(f.w);
      }
    }
    __syncthreads();

    // ---- build fragments per ISA VGPR layouts, 4 WMMAs per wave ----
    // A 16x32 bf16: lanes 0-15 row M=ln hold K 0..7 then 16..23;
    //               lanes 16-31 row M=ln hold K 8..15 then 24..31.
    const int kOffA = half ? 8 : 0;
    Frag16 a[2], b[2];
    #pragma unroll
    for (int mt = 0; mt < 2; ++mt) {
      const unsigned short* s = &ldsA[(waveM * 32 + mt * 16 + ln) * 32];
      a[mt].q[0] = *(const uint4*)&s[kOffA];
      a[mt].q[1] = *(const uint4*)&s[kOffA + 16];
    }
    // B 32x16 bf16: lanes 0-15 col N=ln hold K 0..7 then 16..23; lanes 16-31 K 8..15 then 24..31.
    #pragma unroll
    for (int nt = 0; nt < 2; ++nt) {
      const unsigned short* s = &ldsB[(waveN * 32 + nt * 16 + ln) * 32 + half * 8];
      b[nt].q[0] = *(const uint4*)&s[0];
      b[nt].q[1] = *(const uint4*)&s[16];
    }
    #pragma unroll
    for (int mt = 0; mt < 2; ++mt)
      #pragma unroll
      for (int nt = 0; nt < 2; ++nt)
        acc[mt][nt] = __builtin_amdgcn_wmma_f32_16x16x32_bf16(
            false, a[mt].v, false, b[nt].v, (short)0, acc[mt][nt], false, false);

    __syncthreads();
  }

  // ---- store C (fp32), guard M boundary; N is a multiple of 64 ----
  #pragma unroll
  for (int mt = 0; mt < 2; ++mt) {
    #pragma unroll
    for (int nt = 0; nt < 2; ++nt) {
      const int col = colBase + waveN * 32 + nt * 16 + ln;
      #pragma unroll
      for (int v = 0; v < 8; ++v) {
        const int row = rowBase + waveM * 32 + mt * 16 + half * 8 + v;
        if (row < M) C[(size_t)row * N + col] = acc[mt][nt][v];
      }
    }
  }
}

// --------------------------- degree / norms --------------------------------
__global__ void gcn_deg_init(float* deg, int n) {
  int i = blockIdx.x * blockDim.x + threadIdx.x;
  if (i < n) deg[i] = 1.0f;                    // self-loop weight
}

__global__ void gcn_deg_acc(const long long* __restrict__ ei,
                            const float* __restrict__ w, float* deg, int E) {
  int e = blockIdx.x * blockDim.x + threadIdx.x;
  if (e < E) atomicAdd(&deg[(int)ei[E + e]], w[e]);   // dst row of edge_index
}

__global__ void gcn_dinv(float* deg, float* selfn, int n) {
  int i = blockIdx.x * blockDim.x + threadIdx.x;
  if (i < n) {
    float d  = deg[i];
    float di = (d > 0.f) ? rsqrtf(d) : 0.f;
    deg[i]   = di;                 // deg buffer becomes dinv
    selfn[i] = di * di;            // self-loop norm: dinv[i]*1*dinv[i]
  }
}

// --------------------------- CSR build (counting sort by dst) --------------
__global__ void gcn_zero_u32(unsigned* p, int n) {
  int i = blockIdx.x * blockDim.x + threadIdx.x;
  if (i < n) p[i] = 0u;
}

__global__ void gcn_count(const long long* __restrict__ ei, unsigned* cnt, int E) {
  int e = blockIdx.x * blockDim.x + threadIdx.x;
  if (e < E) atomicAdd(&cnt[(int)ei[E + e]], 1u);
}

// per-block exclusive scan (Hillis-Steele in LDS) + block totals
__global__ __launch_bounds__(256) void gcn_scan1(
    const unsigned* __restrict__ cnt, unsigned* rowptr, unsigned* bsum, int n) {
  __shared__ unsigned lds[256];
  const int t = threadIdx.x;
  const int i = blockIdx.x * 256 + t;
  unsigned v = (i < n) ? cnt[i] : 0u;
  lds[t] = v;
  #pragma unroll
  for (int off = 1; off < 256; off <<= 1) {
    __syncthreads();
    unsigned u = (t >= off) ? lds[t - off] : 0u;
    __syncthreads();
    lds[t] += u;
  }
  if (i < n) rowptr[i] = lds[t] - v;           // exclusive within block
  if (t == 255) bsum[blockIdx.x] = lds[255];   // block total
}

// exclusive scan of block totals (single block; nb <= 256)
__global__ __launch_bounds__(256) void gcn_scan2(
    const unsigned* __restrict__ bsum, unsigned* boff, int nb) {
  __shared__ unsigned lds[256];
  const int t = threadIdx.x;
  unsigned v = (t < nb) ? bsum[t] : 0u;
  lds[t] = v;
  #pragma unroll
  for (int off = 1; off < 256; off <<= 1) {
    __syncthreads();
    unsigned u = (t >= off) ? lds[t - off] : 0u;
    __syncthreads();
    lds[t] += u;
  }
  if (t < nb) boff[t] = lds[t] - v;
}

__global__ void gcn_scan3(unsigned* rowptr, const unsigned* __restrict__ boff,
                          int n, int E) {
  int i = blockIdx.x * blockDim.x + threadIdx.x;
  if (i < n) rowptr[i] += boff[i >> 8];
  if (i == 0) rowptr[n] = (unsigned)E;         // tail sentinel
}

// permute edges into CSR order; precompute per-edge norms
__global__ void gcn_csr_fill(const long long* __restrict__ ei,
                             const float* __restrict__ ea,
                             const float* __restrict__ dinv,
                             const unsigned* __restrict__ rowptr,
                             unsigned* cursor,
                             int* __restrict__ esrc,
                             float* __restrict__ enorm, int E) {
  int e = blockIdx.x * blockDim.x + threadIdx.x;
  if (e < E) {
    int s = (int)ei[e];
    int d = (int)ei[E + e];
    unsigned p = rowptr[d] + atomicAdd(&cursor[d], 1u);
    esrc[p]  = s;
    enorm[p] = dinv[s] * ea[e] * dinv[d];
  }
}

// --------------------------- fused aggregation -----------------------------
// out[i,:] = (relu?) ( Y[i,:]*selfn[i] + bias + sum_{j in CSR(i)} Y[src_j,:]*norm_j )
// One thread = (node i, chunk of 4 features). Gather-only: no atomics.
__global__ void gcn_aggregate(const float* __restrict__ Y,
                              const unsigned* __restrict__ rowptr,
                              const int* __restrict__ esrc,
                              const float* __restrict__ enorm,
                              const float* __restrict__ selfn,
                              const float* __restrict__ bias,
                              float* __restrict__ out,
                              int n, int ld2D, int relu) {
  const int ld2C = ld2D - 2;                    // float4 chunks per row
  const int cm   = (1 << ld2C) - 1;
  long long t = (long long)blockIdx.x * blockDim.x + threadIdx.x;
  if (t >= ((long long)n << ld2C)) return;
  const int i = (int)(t >> ld2C);
  const int c = (int)(t & cm);
  const size_t cpr = (size_t)1 << ld2C;

  const float4* Y4 = (const float4*)Y;
  const float   sn = selfn[i];
  float4 acc = Y4[(size_t)i * cpr + c];
  const float4 bv = ((const float4*)bias)[c];
  acc.x = acc.x * sn + bv.x;
  acc.y = acc.y * sn + bv.y;
  acc.z = acc.z * sn + bv.z;
  acc.w = acc.w * sn + bv.w;

  const unsigned jb = rowptr[i];
  const unsigned je = rowptr[i + 1];
  for (unsigned j = jb; j < je; ++j) {
    const float  w = enorm[j];                  // broadcast across node threads
    const float4 y = Y4[(size_t)esrc[j] * cpr + c];  // coalesced row gather
    acc.x += y.x * w;
    acc.y += y.y * w;
    acc.z += y.z * w;
    acc.w += y.w * w;
  }
  if (relu) {
    acc.x = fmaxf(acc.x, 0.f);
    acc.y = fmaxf(acc.y, 0.f);
    acc.z = fmaxf(acc.z, 0.f);
    acc.w = fmaxf(acc.w, 0.f);
  }
  ((float4*)out)[(size_t)i * cpr + c] = acc;
}

// --------------------------- driver ----------------------------------------
static inline unsigned blks(long long total, int bs) {
  return (unsigned)((total + bs - 1) / bs);
}

extern "C" void kernel_launch(void* const* d_in, const int* in_sizes, int n_in,
                              void* d_out, int out_size, void* d_ws, size_t ws_size,
                              hipStream_t stream) {
  const float*     x  = (const float*)d_in[0];        // [N,256]
  const long long* ei = (const long long*)d_in[1];    // [2,E] int64
  const float*     ea = (const float*)d_in[2];        // [E]
  const float* W1 = (const float*)d_in[3];            // [256,512]
  const float* b1 = (const float*)d_in[4];
  const float* W2 = (const float*)d_in[5];            // [512,256]
  const float* b2 = (const float*)d_in[6];
  const float* W3 = (const float*)d_in[7];            // [256,128]
  const float* b3 = (const float*)d_in[8];
  float* out = (float*)d_out;                         // [N,128]

  const int N = in_sizes[0] / 256;                    // 50000
  const int E = in_sizes[2];                          // 800000

  // workspace layout
  float* ws    = (float*)d_ws;
  float* dinv  = ws;                                  // N  (deg -> dinv)
  float* selfn = dinv + N;                            // N
  float* enorm = selfn + N;                           // E  (CSR-ordered norms)
  float* Y     = enorm + E;                           // N*512
  float* H     = Y + (size_t)N * 512;                 // N*512
  unsigned* rowptr = (unsigned*)(H + (size_t)N * 512);// N+1
  unsigned* cnt    = rowptr + (N + 1);                // N  (counts, then cursor)
  unsigned* bsum   = cnt + N;                         // 256
  unsigned* boff   = bsum + 256;                      // 256
  int*      esrc   = (int*)(boff + 256);              // E  (CSR-ordered sources)

  const int BS = 256;
  const int NB = (N + 255) / 256;                     // scan blocks (<=256)

  // ---- symmetric normalization with self-loops ----
  gcn_deg_init<<<blks(N, BS), BS, 0, stream>>>(dinv, N);
  gcn_deg_acc<<<blks(E, BS), BS, 0, stream>>>(ei, ea, dinv, E);
  gcn_dinv<<<blks(N, BS), BS, 0, stream>>>(dinv, selfn, N);

  // ---- CSR build: count -> scan -> permute (with fused norm compute) ----
  gcn_zero_u32<<<blks(N, BS), BS, 0, stream>>>(cnt, N);
  gcn_count<<<blks(E, BS), BS, 0, stream>>>(ei, cnt, E);
  gcn_scan1<<<NB, 256, 0, stream>>>(cnt, rowptr, bsum, N);
  gcn_scan2<<<1, 256, 0, stream>>>(bsum, boff, NB);
  gcn_scan3<<<blks(N, BS), BS, 0, stream>>>(rowptr, boff, N, E);
  gcn_zero_u32<<<blks(N, BS), BS, 0, stream>>>(cnt, N);
  gcn_csr_fill<<<blks(E, BS), BS, 0, stream>>>(ei, ea, dinv, rowptr, cnt, esrc, enorm, E);

  // ---- layer 1: x[N,256] @ W1 -> Y[N,512]; gather-reduce -> H; relu ----
  {
    const int K = 256, D = 512, ld2D = 9;
    dim3 g((N + 63) / 64, D / 64);
    gcn_gemm_wmma<<<g, 128, 0, stream>>>(x, W1, Y, N, K, D);
    gcn_aggregate<<<blks((long long)N << (ld2D - 2), BS), BS, 0, stream>>>(
        Y, rowptr, esrc, enorm, selfn, b1, H, N, ld2D, 1);
  }

  // ---- layer 2: H[N,512] @ W2 -> Y[N,256]; gather-reduce -> H; relu ----
  {
    const int K = 512, D = 256, ld2D = 8;
    dim3 g((N + 63) / 64, D / 64);
    gcn_gemm_wmma<<<g, 128, 0, stream>>>(H, W2, Y, N, K, D);
    gcn_aggregate<<<blks((long long)N << (ld2D - 2), BS), BS, 0, stream>>>(
        Y, rowptr, esrc, enorm, selfn, b2, H, N, ld2D, 1);
  }

  // ---- layer 3: H[N,256] @ W3 -> Y[N,128]; gather-reduce -> d_out ----
  {
    const int K = 256, D = 128, ld2D = 7;
    dim3 g((N + 63) / 64, D / 64);
    gcn_gemm_wmma<<<g, 128, 0, stream>>>(H, W3, Y, N, K, D);
    gcn_aggregate<<<blks((long long)N << (ld2D - 2), BS), BS, 0, stream>>>(
        Y, rowptr, esrc, enorm, selfn, b3, out, N, ld2D, 0);
  }
}
